// GraphUNet_79517024518474
// MI455X (gfx1250) — matrix operand order repaired and checked by reading, not compiled, not run on hardware
//
#include <hip/hip_runtime.h>
#include <math.h>

typedef __attribute__((ext_vector_type(16))) _Float16 v16h;
typedef __attribute__((ext_vector_type(8)))  float    v8f;

#define NGROUPS 8
#define NHEADS  8

enum { ACT_NONE = 0, ACT_SILU = 1 };
enum { ST_STORE = 0, ST_ADD = 1, ST_SCATTER = 2 };
enum { A_DIRECT = 0, A_EDGE = 1 };

__device__ __forceinline__ float silu_f(float x) { return x / (1.f + __expf(-x)); }

// ---------------------------------------------------------------------------
// WMMA GEMM: Out(N x M) = A(N x K) @ W(K x M) + bias, f16 tiles, f32 acc.
//   - 256 threads = 8 wave32 in a 4x2 grid; wave tile = 32x32 (4 WMMAs/step);
//     workgroup tile = 128 rows x 64 cols; K stepped by 32.
//   - A/B staged into LDS in fragment-swizzled order: every wave fragment is
//     one contiguous 32B read (ds_load_b128 x2), A stage stores ds_store_b128.
//   - AMODE: DIRECT buffer or fused edge-gather [x[dst], x[src], ea]
//     (all Cx are multiples of 8, so an 8-wide K segment never straddles a
//     region boundary -> one branch per segment, float4 global loads).
//   - SMODE: store / accumulate / atomic scatter-mean to Out[dst[row]].
// ---------------------------------------------------------------------------
template <int AMODE, int SMODE>
__global__ __launch_bounds__(256) void k_gemm_wmma(
    const float* __restrict__ A, int lda,
    const float* __restrict__ X, int Cx,
    const int* __restrict__ esrc, const int* __restrict__ edst,
    const float* __restrict__ EAT,
    int Nrows, int K,
    const float* __restrict__ W, const float* __restrict__ bias, int M,
    float* __restrict__ Out,
    const float* __restrict__ invcnt)
{
  // fragment-swizzled tiles: [block16][lane][16 halves]
  __shared__ _Float16 sA[8 * 32 * 16];  // 128 rows x 32 k
  __shared__ _Float16 sB[4 * 32 * 16];  // 32 k x 64 cols

  const int tid  = threadIdx.x;
  const int wave = tid >> 5;
  const int lane = tid & 31;
  const int rowBlk = blockIdx.x * 128;
  const int colBlk = blockIdx.y * 64;
  const int wr = (wave & 3) * 32;   // wave row offset in tile
  const int wc = (wave >> 2) * 32;  // wave col offset in tile

  v8f acc00 = {}, acc01 = {}, acc10 = {}, acc11 = {};

  for (int k0 = 0; k0 < K; k0 += 32) {
    // ---- stage A: 512 segments of 8 halves; thread handles seg tid, tid+256
#pragma unroll
    for (int ss = 0; ss < 2; ++ss) {
      const int s  = tid + ss * 256;
      const int r  = s >> 2;           // 0..127
      const int ks = (s & 3) * 8;      // 0,8,16,24
      const int ge = rowBlk + r;
      const int kk = k0 + ks;
      float f[8];
      const bool inb = (ge < Nrows);
      if (AMODE == A_DIRECT) {
        if (inb && kk + 8 <= K) {
          const float4* p = (const float4*)(A + (size_t)ge * lda + kk);
          float4 a = p[0], b = p[1];
          f[0]=a.x; f[1]=a.y; f[2]=a.z; f[3]=a.w;
          f[4]=b.x; f[5]=b.y; f[6]=b.z; f[7]=b.w;
        } else {
#pragma unroll
          for (int j = 0; j < 8; ++j) {
            int kj = kk + j;
            f[j] = (inb && kj < K) ? A[(size_t)ge * lda + kj] : 0.f;
          }
        }
      } else {
        if (inb && kk + 8 <= Cx) {
          const float4* p = (const float4*)(X + (size_t)edst[ge] * Cx + kk);
          float4 a = p[0], b = p[1];
          f[0]=a.x; f[1]=a.y; f[2]=a.z; f[3]=a.w;
          f[4]=b.x; f[5]=b.y; f[6]=b.z; f[7]=b.w;
        } else if (inb && kk + 8 <= 2 * Cx) {
          const float4* p = (const float4*)(X + (size_t)esrc[ge] * Cx + (kk - Cx));
          float4 a = p[0], b = p[1];
          f[0]=a.x; f[1]=a.y; f[2]=a.z; f[3]=a.w;
          f[4]=b.x; f[5]=b.y; f[6]=b.z; f[7]=b.w;
        } else if (inb) {
          // kk >= 2*Cx here (both multiples of 8): edge-attr tail + zero pad
#pragma unroll
          for (int j = 0; j < 8; ++j) {
            int kj = kk + j;
            f[j] = (kj < K) ? EAT[(size_t)ge * 4 + (kj - 2 * Cx)] : 0.f;
          }
        } else {
#pragma unroll
          for (int j = 0; j < 8; ++j) f[j] = 0.f;
        }
      }
      union { _Float16 h[8]; uint4 u; } pk;
#pragma unroll
      for (int j = 0; j < 8; ++j) pk.h[j] = (_Float16)f[j];
      // swizzled dest: block=(r>>4), lane=hi*16+(r&15), elem base 0/8
      const int hi  = (ks >> 4) & 1;
      const int eb  = (ks & 8) ? 8 : 0;
      const int off = (((r >> 4) * 32) + hi * 16 + (r & 15)) * 16 + eb;
      *(uint4*)&sA[off] = pk.u;  // 16B aligned (off % 8 == 0)
    }
    // ---- stage B: thread -> k row tid>>3, 8 consecutive cols
    {
      const int kr = tid >> 3;          // 0..31
      const int ns = (tid & 7) * 8;     // 0..56
      const int kk = k0 + kr;
      float f[8];
      if (kk < K) {
        const float4* p = (const float4*)(W + (size_t)kk * M + colBlk + ns);
        float4 a = p[0], b = p[1];
        f[0]=a.x; f[1]=a.y; f[2]=a.z; f[3]=a.w;
        f[4]=b.x; f[5]=b.y; f[6]=b.z; f[7]=b.w;
      } else {
#pragma unroll
        for (int j = 0; j < 8; ++j) f[j] = 0.f;
      }
      const int e  = kr & 15;
      const int hb = (kr >> 4) * 16;
#pragma unroll
      for (int j = 0; j < 8; ++j) {
        int n = ns + j;
        sB[((n >> 4) * 32 + hb + (n & 15)) * 16 + e] = (_Float16)f[j];
      }
    }
    __syncthreads();

    // ---- fragment fetch (contiguous 32B per lane) + 4 WMMAs
    const v16h a0 = *(const v16h*)&sA[(((wave & 3) * 2 + 0) * 32 + lane) * 16];
    const v16h a1 = *(const v16h*)&sA[(((wave & 3) * 2 + 1) * 32 + lane) * 16];
    const v16h b0 = *(const v16h*)&sB[(((wave >> 2) * 2 + 0) * 32 + lane) * 16];
    const v16h b1 = *(const v16h*)&sB[(((wave >> 2) * 2 + 1) * 32 + lane) * 16];
    acc00 = __builtin_amdgcn_wmma_f32_16x16x32_f16(false, a0, false, b0, (short)0, acc00, false, false);
    acc01 = __builtin_amdgcn_wmma_f32_16x16x32_f16(false, a0, false, b1, (short)0, acc01, false, false);
    acc10 = __builtin_amdgcn_wmma_f32_16x16x32_f16(false, a1, false, b0, (short)0, acc10, false, false);
    acc11 = __builtin_amdgcn_wmma_f32_16x16x32_f16(false, a1, false, b1, (short)0, acc11, false, false);
    __syncthreads();
  }

  // ---- epilogue: bias + store / add / scatter-mean (M % 64 == 0, no col guard)
  auto epi = [&](const v8f& a, int mBase, int nBase) {
    const int ncol = nBase + (lane & 15);
    const float bv = bias ? bias[ncol] : 0.f;
#pragma unroll
    for (int r = 0; r < 8; ++r) {
      int mrow = mBase + ((lane >> 4) ? 8 : 0) + r;
      if (mrow >= Nrows) continue;
      float v = a[r] + bv;
      if (SMODE == ST_STORE) {
        Out[(size_t)mrow * M + ncol] = v;
      } else if (SMODE == ST_ADD) {
        Out[(size_t)mrow * M + ncol] += v;
      } else {
        int d = edst[mrow];
        atomicAdd(&Out[(size_t)d * M + ncol], v * invcnt[d]);
      }
    }
  };
  epi(acc00, rowBlk + wr,      colBlk + wc);
  epi(acc01, rowBlk + wr,      colBlk + wc + 16);
  epi(acc10, rowBlk + wr + 16, colBlk + wc);
  epi(acc11, rowBlk + wr + 16, colBlk + wc + 16);
}

// ---------------------------------------------------------------------------
// Scalar helpers
// ---------------------------------------------------------------------------
__global__ void k_small_gemm(const float* __restrict__ A, int N, int K,
                             const float* __restrict__ W, const float* __restrict__ b,
                             int M, float* __restrict__ out, int act)
{
  int i = blockIdx.x * blockDim.x + threadIdx.x;
  if (i >= N * M) return;
  int m = i % M, r = i / M;
  float s = b ? b[m] : 0.f;
  for (int k = 0; k < K; ++k) s += A[(size_t)r * K + k] * W[(size_t)k * M + m];
  if (act == ACT_SILU) s = silu_f(s);
  out[i] = s;
}

__global__ void k_groupnorm(float* __restrict__ x, int N, int C,
                            const float* __restrict__ g, const float* __restrict__ b,
                            int fuse_silu)
{
  int i = blockIdx.x * blockDim.x + threadIdx.x;
  if (i >= N * NGROUPS) return;
  int grp = i % NGROUPS, r = i / NGROUPS;
  int cg = C / NGROUPS;
  float* p = x + (size_t)r * C + grp * cg;
  float mu = 0.f;
  for (int j = 0; j < cg; ++j) mu += p[j];
  mu /= cg;
  float var = 0.f;
  for (int j = 0; j < cg; ++j) { float d = p[j] - mu; var += d * d; }
  var /= cg;
  float rs = rsqrtf(var + 1e-5f);
  int cb = grp * cg;
  for (int j = 0; j < cg; ++j) {
    float v = (p[j] - mu) * rs * g[cb + j] + b[cb + j];
    if (fuse_silu) v = silu_f(v);
    p[j] = v;
  }
}

__global__ void k_count(const int* __restrict__ dst, int E, float* __restrict__ cnt)
{
  int e = blockIdx.x * blockDim.x + threadIdx.x;
  if (e < E) atomicAdd(&cnt[dst[e]], 1.f);
}

__global__ void k_invcnt(const float* __restrict__ cnt, float* __restrict__ inv, int n)
{
  int i = blockIdx.x * blockDim.x + threadIdx.x;
  if (i < n) inv[i] = 1.f / fmaxf(cnt[i], 1.f);
}

__global__ void k_time_emb(const int* __restrict__ t, float* __restrict__ emb, int B)
{
  int i = blockIdx.x * blockDim.x + threadIdx.x;
  if (i >= B * 128) return;
  int b = i / 128, c = i % 128;
  float lg = logf(10000.f);
  float arg;
  if (c < 64) { arg = (float)t[b] * __expf(-lg * (float)c / 64.f);        emb[i] = sinf(arg); }
  else        { arg = (float)t[b] * __expf(-lg * (float)(c - 64) / 64.f); emb[i] = cosf(arg); }
}

__global__ void k_silu_copy(const float* __restrict__ in, float* __restrict__ out, int n)
{
  int i = blockIdx.x * blockDim.x + threadIdx.x;
  if (i < n) out[i] = silu_f(in[i]);
}

__global__ void k_badd(float* __restrict__ h, const float* __restrict__ tp,
                       int N, int C, int V)
{
  long i = (long)blockIdx.x * blockDim.x + threadIdx.x;
  if (i >= (long)N * C) return;
  int r = (int)(i / C), c = (int)(i % C);
  h[i] += tp[(r / V) * C + c];
}

__global__ void k_add(float* __restrict__ a, const float* __restrict__ b, long n)
{
  long i = (long)blockIdx.x * blockDim.x + threadIdx.x;
  if (i < n) a[i] += b[i];
}

__global__ void k_concat(const float* __restrict__ a, int Ca,
                         const float* __restrict__ b, int Cb,
                         float* __restrict__ o, int N)
{
  int C = Ca + Cb;
  long i = (long)blockIdx.x * blockDim.x + threadIdx.x;
  if (i >= (long)N * C) return;
  int r = (int)(i / C), c = (int)(i % C);
  o[i] = (c < Ca) ? a[(size_t)r * Ca + c] : b[(size_t)r * Cb + (c - Ca)];
}

// ---------------------------------------------------------------------------
// Attention (C=256, H=8, hd=32)
// ---------------------------------------------------------------------------
__device__ __forceinline__ unsigned f2o(float f) {
  unsigned u = __float_as_uint(f);
  return (u & 0x80000000u) ? ~u : (u | 0x80000000u);
}
__device__ __forceinline__ float o2f(unsigned u) {
  return (u & 0x80000000u) ? __uint_as_float(u ^ 0x80000000u) : __uint_as_float(~u);
}

__global__ void k_edge_score(const float* __restrict__ q, const float* __restrict__ k,
                             const float* __restrict__ ea,
                             const float* __restrict__ we, const float* __restrict__ be,
                             const int* __restrict__ src, const int* __restrict__ dst,
                             int E, float* __restrict__ sc)
{
  int i = blockIdx.x * blockDim.x + threadIdx.x;
  if (i >= E * NHEADS) return;
  int h = i % NHEADS, e = i / NHEADS;
  const float* qp = q + (size_t)dst[e] * 256 + h * 32;
  const float* kp = k + (size_t)src[e] * 256 + h * 32;
  float s = 0.f;
  for (int j = 0; j < 32; ++j) s += qp[j] * kp[j];
  s *= 0.17677669529663689f; // 32^-0.5
  float el = be[h];
  for (int j = 0; j < 4; ++j) el += ea[(size_t)e * 4 + j] * we[j * NHEADS + h];
  sc[i] = s + el;
}

__global__ void k_segmax(const float* __restrict__ sc, const int* __restrict__ dst,
                         int E, unsigned* __restrict__ amax)
{
  int i = blockIdx.x * blockDim.x + threadIdx.x;
  if (i >= E * NHEADS) return;
  int h = i % NHEADS, e = i / NHEADS;
  atomicMax(&amax[(size_t)dst[e] * NHEADS + h], f2o(sc[i]));
}

__global__ void k_expsum(float* __restrict__ sc, const int* __restrict__ dst,
                         const unsigned* __restrict__ amax, float* __restrict__ asum,
                         int E)
{
  int i = blockIdx.x * blockDim.x + threadIdx.x;
  if (i >= E * NHEADS) return;
  int h = i % NHEADS, e = i / NHEADS;
  float a = __expf(sc[i] - o2f(amax[(size_t)dst[e] * NHEADS + h]));
  sc[i] = a;
  atomicAdd(&asum[(size_t)dst[e] * NHEADS + h], a);
}

__global__ void k_attn_scatter(const float* __restrict__ sc, const float* __restrict__ v,
                               const int* __restrict__ src, const int* __restrict__ dst,
                               const float* __restrict__ asum, float* __restrict__ out,
                               int E)
{
  int i = blockIdx.x * blockDim.x + threadIdx.x;
  if (i >= E * NHEADS) return;
  int h = i % NHEADS, e = i / NHEADS;
  float a = sc[i] / fmaxf(asum[(size_t)dst[e] * NHEADS + h], 1e-6f);
  const float* vp = v + (size_t)src[e] * 256 + h * 32;
  float* op = out + (size_t)dst[e] * 256 + h * 32;
  for (int j = 0; j < 32; ++j) atomicAdd(&op[j], a * vp[j]);
}

// ---------------------------------------------------------------------------
// Host-side orchestration
// ---------------------------------------------------------------------------
namespace {

struct LinP { const float* w; const float* b; };
struct GnP  { const float* g; const float* b; };
struct MCp  { LinP l1; GnP gn; LinP l2; };
struct ResP { MCp c1, c2; LinP tm; GnP n1, n2; LinP skip; int has_skip; int ci, co; };
struct AttnP{ ResP res; LinP q, k, v, e, o; GnP norm; };

struct Ctx {
  hipStream_t s;
  int n, B, V, E;
  const int *src, *dst;
  const float* ea;
  const float* stemb;   // (B,256) silu(temb)
  float* tproj;         // (B,512) scratch
  float* EB;            // E x 256 edge intermediate
  float *QB, *KB, *VB;  // n x 256
  float* SC;            // E x 8
  unsigned* AMAX;       // n x 8
  float* ASUM;          // n x 8
  float* INV;           // n
};

static inline int ceil_div(long a, int b) { return (int)((a + b - 1) / b); }

static void gemm_wmma(hipStream_t s, int amode, int smode,
                      const float* A, int lda,
                      const float* X, int Cx,
                      const int* esrc, const int* edst, const float* EAT,
                      int N, int K, const float* W, const float* b, int M,
                      float* out, const float* invcnt)
{
  dim3 grid((N + 127) / 128, M / 64);
  if (amode == A_EDGE) {
    k_gemm_wmma<A_EDGE, ST_STORE><<<grid, 256, 0, s>>>(
        A, lda, X, Cx, esrc, edst, EAT, N, K, W, b, M, out, invcnt);
  } else if (smode == ST_STORE) {
    k_gemm_wmma<A_DIRECT, ST_STORE><<<grid, 256, 0, s>>>(
        A, lda, X, Cx, esrc, edst, EAT, N, K, W, b, M, out, invcnt);
  } else if (smode == ST_ADD) {
    k_gemm_wmma<A_DIRECT, ST_ADD><<<grid, 256, 0, s>>>(
        A, lda, X, Cx, esrc, edst, EAT, N, K, W, b, M, out, invcnt);
  } else {
    k_gemm_wmma<A_DIRECT, ST_SCATTER><<<grid, 256, 0, s>>>(
        A, lda, X, Cx, esrc, edst, EAT, N, K, W, b, M, out, invcnt);
  }
}

static void dense(hipStream_t s, const float* A, int N, int K,
                  const float* W, const float* b, int M, float* out,
                  int act, int smode, const int* edst, const float* invcnt)
{
  if (M % 64 == 0 && K >= 16 && act == ACT_NONE) {
    gemm_wmma(s, A_DIRECT, smode, A, K, nullptr, 0, nullptr, edst, nullptr,
              N, K, W, b, M, out, invcnt);
  } else {
    k_small_gemm<<<ceil_div((long)N * M, 256), 256, 0, s>>>(A, N, K, W, b, M, out, act);
  }
}

static void mesh_conv(Ctx& c, const MCp& p, const float* xin, int Cin, int Cout,
                      float* out_node)
{
  hipMemsetAsync(out_node, 0, (size_t)c.n * Cout * sizeof(float), c.s);
  // l1: fused gather [x[dst],x[src],ea] -> E x Cout
  gemm_wmma(c.s, A_EDGE, ST_STORE, nullptr, 0, xin, Cin, c.src, c.dst, c.ea,
            c.E, 2 * Cin + 4, p.l1.w, p.l1.b, Cout, c.EB, nullptr);
  // gn + silu in place on edge rows
  k_groupnorm<<<ceil_div((long)c.E * NGROUPS, 256), 256, 0, c.s>>>(
      c.EB, c.E, Cout, p.gn.g, p.gn.b, 1);
  // l2 with fused scatter-mean epilogue
  gemm_wmma(c.s, A_DIRECT, ST_SCATTER, c.EB, Cout, nullptr, 0, nullptr, c.dst, nullptr,
            c.E, Cout, p.l2.w, p.l2.b, Cout, out_node, c.INV);
}

static void res_block(Ctx& c, const ResP& p, const float* xin, float* out, float* tmp)
{
  mesh_conv(c, p.c1, xin, p.ci, p.co, tmp);
  k_groupnorm<<<ceil_div((long)c.n * NGROUPS, 256), 256, 0, c.s>>>(
      tmp, c.n, p.co, p.n1.g, p.n1.b, 1);
  // time projection (B x co), broadcast add
  k_small_gemm<<<ceil_div((long)c.B * p.co, 256), 256, 0, c.s>>>(
      c.stemb, c.B, 256, p.tm.w, p.tm.b, p.co, c.tproj, ACT_NONE);
  k_badd<<<ceil_div((long)c.n * p.co, 256), 256, 0, c.s>>>(tmp, c.tproj, c.n, p.co, c.V);
  mesh_conv(c, p.c2, tmp, p.co, p.co, out);
  k_groupnorm<<<ceil_div((long)c.n * NGROUPS, 256), 256, 0, c.s>>>(
      out, c.n, p.co, p.n2.g, p.n2.b, 1);
  if (p.has_skip)
    dense(c.s, xin, c.n, p.ci, p.skip.w, p.skip.b, p.co, out,
          ACT_NONE, ST_ADD, nullptr, nullptr);
  else
    k_add<<<ceil_div((long)c.n * p.co, 256), 256, 0, c.s>>>(out, xin, (long)c.n * p.co);
}

static void attn_block(Ctx& c, const AttnP& p, const float* xin,
                       float* out, float* t1, float* t2)
{
  res_block(c, p.res, xin, out, t1); // out = hres (n x 256)
  const int D = 256;
  dense(c.s, out, c.n, D, p.q.w, p.q.b, D, c.QB, ACT_NONE, ST_STORE, nullptr, nullptr);
  dense(c.s, out, c.n, D, p.k.w, p.k.b, D, c.KB, ACT_NONE, ST_STORE, nullptr, nullptr);
  dense(c.s, out, c.n, D, p.v.w, p.v.b, D, c.VB, ACT_NONE, ST_STORE, nullptr, nullptr);
  hipMemsetAsync(c.AMAX, 0, (size_t)c.n * NHEADS * sizeof(unsigned), c.s);
  hipMemsetAsync(c.ASUM, 0, (size_t)c.n * NHEADS * sizeof(float), c.s);
  int eh = c.E * NHEADS;
  k_edge_score<<<ceil_div(eh, 256), 256, 0, c.s>>>(c.QB, c.KB, c.ea, p.e.w, p.e.b,
                                                   c.src, c.dst, c.E, c.SC);
  k_segmax<<<ceil_div(eh, 256), 256, 0, c.s>>>(c.SC, c.dst, c.E, c.AMAX);
  k_expsum<<<ceil_div(eh, 256), 256, 0, c.s>>>(c.SC, c.dst, c.AMAX, c.ASUM, c.E);
  hipMemsetAsync(t2, 0, (size_t)c.n * D * sizeof(float), c.s);
  k_attn_scatter<<<ceil_div(eh, 256), 256, 0, c.s>>>(c.SC, c.VB, c.src, c.dst,
                                                     c.ASUM, t2, c.E);
  dense(c.s, t2, c.n, D, p.o.w, p.o.b, D, t1, ACT_NONE, ST_STORE, nullptr, nullptr);
  k_groupnorm<<<ceil_div((long)c.n * NGROUPS, 256), 256, 0, c.s>>>(
      t1, c.n, D, p.norm.g, p.norm.b, 0);
  k_add<<<ceil_div((long)c.n * D, 256), 256, 0, c.s>>>(out, t1, (long)c.n * D);
}

} // namespace

extern "C" void kernel_launch(void* const* d_in, const int* in_sizes, int n_in,
                              void* d_out, int out_size, void* d_ws, size_t ws_size,
                              hipStream_t stream)
{
  const float* x  = (const float*)d_in[0];
  const int*   t  = (const int*)d_in[1];
  const int*   ei = (const int*)d_in[2];
  const float* ea = (const float*)d_in[3];

  const int n = in_sizes[0] / 3;      // B*V nodes
  const int B = in_sizes[1];
  const int V = n / B;
  const int E = in_sizes[2] / 2;
  const int* src = ei;
  const int* dst = ei + E;

  // ---- parameter cursor (setup_inputs insertion order, depth-first) ----
  int gi = 4;
  auto nxt   = [&]() { return (const float*)d_in[gi++]; };
  auto rdLin = [&]() { LinP l; l.w = nxt(); l.b = nxt(); return l; };
  auto rdGN  = [&]() { GnP g; g.g = nxt(); g.b = nxt(); return g; };
  auto rdMC  = [&]() { MCp m; m.l1 = rdLin(); m.gn = rdGN(); m.l2 = rdLin(); return m; };
  auto rdRes = [&](int ci, int co) {
    ResP r; r.ci = ci; r.co = co;
    r.c1 = rdMC(); r.c2 = rdMC(); r.tm = rdLin(); r.n1 = rdGN(); r.n2 = rdGN();
    r.has_skip = (ci != co);
    if (r.has_skip) r.skip = rdLin(); else r.skip = LinP{nullptr, nullptr};
    return r;
  };
  auto rdAttn = [&](int d) {
    AttnP a; a.res = rdRes(d, d);
    a.q = rdLin(); a.k = rdLin(); a.v = rdLin();
    a.e = rdLin(); a.o = rdLin(); a.norm = rdGN();
    return a;
  };

  LinP time_l1 = rdLin();
  LinP time_l2 = rdLin();
  LinP inp     = rdLin();
  ResP e00 = rdRes(64, 64),   e01 = rdRes(64, 64);
  ResP e10 = rdRes(64, 128),  e11 = rdRes(128, 128);
  LinP e20lin = rdLin();      AttnP e20 = rdAttn(256);
  AttnP e21 = rdAttn(256);
  AttnP e30 = rdAttn(256),    e31 = rdAttn(256);
  AttnP mid1 = rdAttn(256),   mid2 = rdAttn(256);
  LinP d00lin = rdLin();      AttnP d00 = rdAttn(256);
  AttnP d01 = rdAttn(256);
  LinP d10lin = rdLin();      AttnP d10 = rdAttn(256);
  AttnP d11 = rdAttn(256);
  ResP d20 = rdRes(384, 128), d21 = rdRes(128, 128);
  ResP d30 = rdRes(192, 64),  d31 = rdRes(64, 64);
  GnP out_norm = rdGN();
  LinP outW    = rdLin();

  // ---- workspace layout ----
  char* w8 = (char*)d_ws;
  size_t off = 0;
  auto alloc = [&](size_t elems) -> float* {
    float* p = (float*)(w8 + off);
    off += ((elems * sizeof(float) + 255) / 256) * 256;
    return p;
  };
  float* NB[4];
  for (int i = 0; i < 4; ++i) NB[i] = alloc((size_t)n * 512);
  float* SK0 = alloc((size_t)n * 64);
  float* SK1 = alloc((size_t)n * 128);
  float* SK2 = alloc((size_t)n * 256);
  float* SK3 = alloc((size_t)n * 256);

  Ctx c;
  c.s = stream; c.n = n; c.B = B; c.V = V; c.E = E;
  c.src = src; c.dst = dst; c.ea = ea;
  c.EB = alloc((size_t)E * 256);
  c.QB = alloc((size_t)n * 256);
  c.KB = alloc((size_t)n * 256);
  c.VB = alloc((size_t)n * 256);
  c.SC = alloc((size_t)E * NHEADS);
  c.AMAX = (unsigned*)alloc((size_t)n * NHEADS);
  c.ASUM = alloc((size_t)n * NHEADS);
  c.INV  = alloc(n);
  float* CNT  = alloc(n);
  float* T128 = alloc((size_t)B * 128);
  float* TA   = alloc((size_t)B * 256);
  float* TEMB = alloc((size_t)B * 256);
  float* STE  = alloc((size_t)B * 256);
  c.tproj = alloc((size_t)B * 512);
  c.stemb = STE;

  // ---- degree counts (mean-scatter scale) ----
  hipMemsetAsync(CNT, 0, (size_t)n * sizeof(float), stream);
  k_count<<<ceil_div(E, 256), 256, 0, stream>>>(dst, E, CNT);
  k_invcnt<<<ceil_div(n, 256), 256, 0, stream>>>(CNT, c.INV, n);

  // ---- time embedding ----
  k_time_emb<<<ceil_div(B * 128, 256), 256, 0, stream>>>(t, T128, B);
  k_small_gemm<<<ceil_div(B * 256, 256), 256, 0, stream>>>(T128, B, 128,
      time_l1.w, time_l1.b, 256, TA, ACT_SILU);
  k_small_gemm<<<ceil_div(B * 256, 256), 256, 0, stream>>>(TA, B, 256,
      time_l2.w, time_l2.b, 256, TEMB, ACT_NONE);
  k_silu_copy<<<ceil_div(B * 256, 256), 256, 0, stream>>>(TEMB, STE, B * 256);

  // ---- input projection (K=3 -> scalar path) ----
  k_small_gemm<<<ceil_div((long)n * 64, 256), 256, 0, stream>>>(x, n, 3,
      inp.w, inp.b, 64, NB[0], ACT_NONE);

  int cur = 0;
  auto IN = [&]() { return NB[cur]; };
  auto O1 = [&]() { return NB[(cur + 1) & 3]; };
  auto O2 = [&]() { return NB[(cur + 2) & 3]; };
  auto O3 = [&]() { return NB[(cur + 3) & 3]; };
  auto adv = [&]() { cur = (cur + 1) & 3; };

  // ---- encoder ----
  res_block(c, e00, IN(), O1(), O2()); adv();
  res_block(c, e01, IN(), O1(), O2()); adv();
  hipMemcpyAsync(SK0, IN(), (size_t)n * 64 * sizeof(float),
                 hipMemcpyDeviceToDevice, stream);

  res_block(c, e10, IN(), O1(), O2()); adv();
  res_block(c, e11, IN(), O1(), O2()); adv();
  hipMemcpyAsync(SK1, IN(), (size_t)n * 128 * sizeof(float),
                 hipMemcpyDeviceToDevice, stream);

  dense(stream, IN(), n, 128, e20lin.w, e20lin.b, 256, O1(),
        ACT_NONE, ST_STORE, nullptr, nullptr); adv();
  attn_block(c, e20, IN(), O1(), O2(), O3()); adv();
  attn_block(c, e21, IN(), O1(), O2(), O3()); adv();
  hipMemcpyAsync(SK2, IN(), (size_t)n * 256 * sizeof(float),
                 hipMemcpyDeviceToDevice, stream);

  attn_block(c, e30, IN(), O1(), O2(), O3()); adv();
  attn_block(c, e31, IN(), O1(), O2(), O3()); adv();
  hipMemcpyAsync(SK3, IN(), (size_t)n * 256 * sizeof(float),
                 hipMemcpyDeviceToDevice, stream);

  // ---- middle ----
  attn_block(c, mid1, IN(), O1(), O2(), O3()); adv();
  attn_block(c, mid2, IN(), O1(), O2(), O3()); adv();

  // ---- decoder ----
  k_concat<<<ceil_div((long)n * 512, 256), 256, 0, stream>>>(IN(), 256, SK3, 256, O1(), n); adv();
  dense(stream, IN(), n, 512, d00lin.w, d00lin.b, 256, O1(),
        ACT_NONE, ST_STORE, nullptr, nullptr); adv();
  attn_block(c, d00, IN(), O1(), O2(), O3()); adv();
  attn_block(c, d01, IN(), O1(), O2(), O3()); adv();

  k_concat<<<ceil_div((long)n * 512, 256), 256, 0, stream>>>(IN(), 256, SK2, 256, O1(), n); adv();
  dense(stream, IN(), n, 512, d10lin.w, d10lin.b, 256, O1(),
        ACT_NONE, ST_STORE, nullptr, nullptr); adv();
  attn_block(c, d10, IN(), O1(), O2(), O3()); adv();
  attn_block(c, d11, IN(), O1(), O2(), O3()); adv();

  k_concat<<<ceil_div((long)n * 384, 256), 256, 0, stream>>>(IN(), 256, SK1, 128, O1(), n); adv();
  res_block(c, d20, IN(), O1(), O2()); adv();
  res_block(c, d21, IN(), O1(), O2()); adv();

  k_concat<<<ceil_div((long)n * 192, 256), 256, 0, stream>>>(IN(), 128, SK0, 64, O1(), n); adv();
  res_block(c, d30, IN(), O1(), O2()); adv();
  res_block(c, d31, IN(), O1(), O2()); adv();

  // ---- output head ----
  k_groupnorm<<<ceil_div((long)n * NGROUPS, 256), 256, 0, stream>>>(
      IN(), n, 64, out_norm.g, out_norm.b, 1);
  k_small_gemm<<<ceil_div((long)n * 3, 256), 256, 0, stream>>>(
      IN(), n, 64, outW.w, outW.b, 3, (float*)d_out, ACT_NONE);
}
